// RNN_78812649882434
// MI455X (gfx1250) — compile-verified, matrix operand order checked by reading, not run on hardware
//
#include <hip/hip_runtime.h>
#include <hip/hip_bf16.h>

// MI455X (gfx1250) LSTM forward.
//   ws layout (needs ~318 MB):
//     [0,            16MiB)      Wt   : bf16 W^T  [4096 x 2048]  (B operand, row = output col)
//     [16MiB,   16+256MiB)       G    : f32  x@Wx+b  [16384 x 4096]
//     [+0,        +256KiB)       hbuf : bf16 ping-pong h  [2][64 x 1024]
//     [+256KiB,  +256KiB+64)     bar  : u64 grid-barrier counter
//     [+64,        +32MiB)       xbf  : bf16 x  [16384 x 1024]
//
// Design: bf16 WMMA (f32 acc); x and W pre-converted/transposed to bf16 once
// (coalesced, out of the GEMM hot loops); time-parallel x@Wx as one GEMM with
// 2x4 tiles/wave; sequential h@Wh in a 32-WG persistent kernel with the Wh
// slab TDM-loaded into LDS (padded for bank-conflict-free ds reads) and a
// per-step grid barrier.

typedef __attribute__((ext_vector_type(16))) __bf16        v16bf;
typedef __attribute__((ext_vector_type(8)))  float         v8f;
typedef __attribute__((ext_vector_type(4)))  unsigned int  ui4;
typedef __attribute__((ext_vector_type(8)))  int           i8v;
typedef __attribute__((ext_vector_type(4)))  int           i4v;

struct __align__(16) U4 { unsigned int x, y, z, w; };
struct __align__(8)  U2 { unsigned int x, y; };
struct __align__(16) F4 { float x, y, z, w; };

union V16 {                    // one WMMA 16-bit operand (A or B), 32 bytes
    v16bf          v;
    unsigned short s[16];
    U4             q[2];
    U2             d[4];
};

#define T_STEPS 256
#define BATCH   64
#define HID     1024
#define KTOT    2048           // I + H
#define N4H     4096
#define MROWS   (T_STEPS * BATCH)   // 16384
#define REC_BLOCKS 32          // 32 WGs x 32 H-cols = 1024
#define GATE_LDS_STRIDE 66048  // 32 rows * 2048B + 64 pads * 8B
#define REC_LDS (4 * GATE_LDS_STRIDE)

__device__ __forceinline__ unsigned short f2bf(float f) {  // RNE f32 -> bf16
    unsigned u = __builtin_bit_cast(unsigned, f);
    u += 0x7FFFu + ((u >> 16) & 1u);
    return (unsigned short)(u >> 16);
}
__device__ __forceinline__ float sigf(float x) { return 1.0f / (1.0f + __expf(-x)); }

// ------------------------------------------- elementwise bf16 conversion ----
// 4 f32 -> 4 bf16 per thread; also h0 -> hbuf[0] and barrier reset.
__global__ __launch_bounds__(256) void lstm_conv(const float* __restrict__ x,
                                                 const float* __restrict__ h0,
                                                 unsigned short* __restrict__ xbf,
                                                 unsigned short* __restrict__ hbuf,
                                                 unsigned long long* __restrict__ bar) {
    size_t tid = (size_t)blockIdx.x * blockDim.x + threadIdx.x;  // 4,194,304 threads
    size_t i   = tid * 4;
    F4 f = *(const F4*)(x + i);
    U2 o;
    o.x = (unsigned)f2bf(f.x) | ((unsigned)f2bf(f.y) << 16);
    o.y = (unsigned)f2bf(f.z) | ((unsigned)f2bf(f.w) << 16);
    *(U2*)(xbf + i) = o;
    if (tid < (BATCH * HID / 4)) {
        F4 h = *(const F4*)(h0 + i);
        U2 ho;
        ho.x = (unsigned)f2bf(h.x) | ((unsigned)f2bf(h.y) << 16);
        ho.y = (unsigned)f2bf(h.z) | ((unsigned)f2bf(h.w) << 16);
        *(U2*)(hbuf + i) = ho;
    }
    if (tid == 0) *bar = 0ull;
}

// --------------------------------- W transpose+convert via LDS 64x64 tile ---
// grid = (4096/64, 2048/64). Coalesced reads along n, coalesced writes along k.
__global__ __launch_bounds__(256) void lstm_wt(const float* __restrict__ W,
                                               unsigned short* __restrict__ Wt) {
    __shared__ unsigned short tile[64][72];          // +8 pad vs LDS banks
    const int n0 = blockIdx.x * 64;
    const int k0 = blockIdx.y * 64;
    const int tn = threadIdx.x & 63;                 // n within tile (coalesced)
    const int tk = threadIdx.x >> 6;                 // k phase 0..3
#pragma unroll
    for (int j = 0; j < 16; ++j) {
        int kl = tk + j * 4;
        tile[tn][kl] = f2bf(W[(size_t)(k0 + kl) * N4H + n0 + tn]);
    }
    __syncthreads();
    const int nl = threadIdx.x >> 2;                 // 0..63: Wt row
    const int kc = (threadIdx.x & 3) * 16;           // 16 contiguous k per thread
    unsigned short* dst = Wt + (size_t)(n0 + nl) * KTOT + k0 + kc;
#pragma unroll
    for (int j = 0; j < 16; ++j) dst[j] = tile[nl][kc + j];
}

// --------------------------------------------------- parallel x @ Wx + b ----
// grid = (4096/64, 16384/256), block = 256. wave = 32x64 strip (2x4 WMMA tiles).
__global__ __launch_bounds__(256) void lstm_xw(const unsigned short* __restrict__ xbf,
                                               const unsigned short* __restrict__ Wt,
                                               const float* __restrict__ bias,
                                               float* __restrict__ G) {
    const int lane = threadIdx.x & 31;
    const int wid  = threadIdx.x >> 5;
    const int r    = lane & 15;
    const int half = lane >> 4;
    const int m0   = blockIdx.y * 256 + wid * 32;   // two row tiles m0, m0+16
    const int n0   = blockIdx.x * 64;               // four col tiles

    v8f acc[2][4] = {{v8f{}, v8f{}, v8f{}, v8f{}}, {v8f{}, v8f{}, v8f{}, v8f{}}};
    const unsigned short* ar0 = xbf + (size_t)(m0 + r) * 1024;
    const unsigned short* ar1 = xbf + (size_t)(m0 + 16 + r) * 1024;

    for (int k0 = 0; k0 < 1024; k0 += 32) {
        V16 a0, a1;
        const unsigned short* p0 = ar0 + k0 + half * 8;
        const unsigned short* p1 = ar1 + k0 + half * 8;
        a0.q[0] = *(const U4*)(p0);  a0.q[1] = *(const U4*)(p0 + 16);
        a1.q[0] = *(const U4*)(p1);  a1.q[1] = *(const U4*)(p1 + 16);
#pragma unroll
        for (int j = 0; j < 4; ++j) {
            V16 bm;   // B tile from bf16 W^T: column (= Wt row) n0+j*16+r
            const unsigned short* wp =
                Wt + (size_t)(n0 + j * 16 + r) * KTOT + k0 + half * 8;
            bm.q[0] = *(const U4*)(wp);
            bm.q[1] = *(const U4*)(wp + 16);
            acc[0][j] = __builtin_amdgcn_wmma_f32_16x16x32_bf16(
                false, a0.v, false, bm.v, (short)0, acc[0][j], false, false);
            acc[1][j] = __builtin_amdgcn_wmma_f32_16x16x32_bf16(
                false, a1.v, false, bm.v, (short)0, acc[1][j], false, false);
        }
    }
#pragma unroll
    for (int p = 0; p < 2; ++p) {
#pragma unroll
        for (int j = 0; j < 4; ++j) {
            int col = n0 + j * 16 + r;
            float bv = bias[col];
#pragma unroll
            for (int v = 0; v < 8; ++v) {
                int m = m0 + p * 16 + half * 8 + v;    // D layout: M = v + 8*half
                G[(size_t)m * N4H + col] = acc[p][j][v] + bv;
            }
        }
    }
}

// -------------------------------------------- sequential recurrence ---------
// grid = 32 WGs (persistent), block = 256 (8 waves), dynamic LDS = REC_LDS.
__global__ __launch_bounds__(256, 1) void lstm_rec(const float* __restrict__ G,
                                                   const float* __restrict__ h0,
                                                   const float* __restrict__ c0,
                                                   const int* __restrict__ L,
                                                   const unsigned short* __restrict__ Wt,
                                                   unsigned short* __restrict__ hbuf,
                                                   float* __restrict__ out,
                                                   unsigned long long* __restrict__ bar) {
    extern __shared__ __align__(16) char smem[];      // Wh slab: 4 gate blocks
    const int lane = threadIdx.x & 31;
    const int wid  = threadIdx.x >> 5;
    const int r    = lane & 15;
    const int half = lane >> 4;
    const int mt   = wid & 3;          // batch tile  0..3  (4x16 = 64 rows)
    const int nt   = wid >> 2;         // col tile    0..1  (2x16 = 32 cols)
    const int n0   = blockIdx.x * 32;  // this WG's H-column base
    const int colh = n0 + nt * 16 + r;

    // ---- TDM: DMA Wh slab (K-major rows, 2-dword pad per 1KiB) into LDS ----
    if (wid == 0) {
#pragma unroll
        for (int g = 0; g < 4; ++g) {
            unsigned lds = (unsigned)(g * GATE_LDS_STRIDE);
            unsigned long long ga = (unsigned long long)(size_t)Wt +
                ((unsigned long long)(g * HID + n0) * KTOT + HID) * 2ull;
            ui4 d0; i8v d1;
            i4v dz4 = {0, 0, 0, 0};
            i8v dz8 = {0, 0, 0, 0, 0, 0, 0, 0};
            d0[0] = 1u;                                          // count=1, user mode
            d0[1] = lds;                                         // lds_addr
            d0[2] = (unsigned)(ga & 0xffffffffull);              // global_addr lo
            d0[3] = (unsigned)((ga >> 32) & 0x1ffffffu) | (2u << 30);  // hi | type=2
            d1[0] = (int)((1u << 16) |      // data_size = 2B
                          (1u << 20) |      // pad_enable
                          (7u << 22) |      // pad_interval: 256 dwords (1 KiB)
                          (1u << 25));      // pad_amount: 2 dwords (8 B)
            d1[1] = (int)(1024u << 16);     // tensor_dim0 = 1024 (K elems)
            d1[2] = (int)(32u << 16);       // tensor_dim1 = 32 rows
            d1[3] = (int)(1024u << 16);     // tile_dim0   = 1024
            d1[4] = (int)(32u);             // tile_dim1   = 32, tile_dim2 = 0
            d1[5] = (int)(2048u);           // tensor_dim0_stride = 2048 (Wt pitch)
            d1[6] = 0; d1[7] = 0;
            __builtin_amdgcn_tensor_load_to_lds(d0, d1, dz4, dz4, dz8, 0);
        }
        __builtin_amdgcn_s_wait_tensorcnt(0);
    }
    __syncthreads();

    // ---- persistent per-lane state: c, frozen h, lengths (8 rows each) ----
    float c[8], hprev[8];
    int   Lr[8];
#pragma unroll
    for (int v = 0; v < 8; ++v) {
        int m    = mt * 16 + half * 8 + v;            // D-layout batch row
        c[v]     = c0[m * HID + colh];
        hprev[v] = h0[m * HID + colh];
        Lr[v]    = L[m];
    }

    for (int t = 0; t < T_STEPS; ++t) {
        const unsigned short* hsrc = hbuf + (size_t)(t & 1) * (BATCH * HID);
        unsigned short*       hdst = hbuf + (size_t)((t + 1) & 1) * (BATCH * HID);

        v8f acc[4] = {v8f{}, v8f{}, v8f{}, v8f{}};    // i, f, o, g accumulators
        const unsigned short* arow = hsrc + (mt * 16 + r) * HID;
        for (int k0 = 0; k0 < 1024; k0 += 32) {
            V16 a;                                     // A = h_t tile (bf16)
            const unsigned short* ap = arow + k0 + half * 8;
            a.q[0] = *(const U4*)(ap);
            a.q[1] = *(const U4*)(ap + 16);
#pragma unroll
            for (int g = 0; g < 4; ++g) {
                V16 bm;                                // B from padded LDS slab
                int xb  = (nt * 16 + r) * 2048 + (k0 + half * 8) * 2;
                int xb2 = xb + 32;
                int p0  = xb  + ((xb  >> 10) << 3);    // +8B per 1KiB of data
                int p1  = xb2 + ((xb2 >> 10) << 3);
                const char* base = smem + g * GATE_LDS_STRIDE;
                bm.d[0] = *(const U2*)(base + p0);
                bm.d[1] = *(const U2*)(base + p0 + 8);
                bm.d[2] = *(const U2*)(base + p1);
                bm.d[3] = *(const U2*)(base + p1 + 8);
                acc[g] = __builtin_amdgcn_wmma_f32_16x16x32_bf16(
                    false, a.v, false, bm.v, (short)0, acc[g], false, false);
            }
        }

        // ---- gate math (pure in-lane thanks to shared D layout) ----
        const float* gp = G + (size_t)t * BATCH * N4H;
#pragma unroll
        for (int v = 0; v < 8; ++v) {
            int m = mt * 16 + half * 8 + v;
            const float* gr = gp + (size_t)m * N4H;
            float iv = acc[0][v] + gr[0 * HID + colh];
            float fv = acc[1][v] + gr[1 * HID + colh];
            float ov = acc[2][v] + gr[2 * HID + colh];
            float gv = acc[3][v] + gr[3 * HID + colh];
            float cn = sigf(iv) * tanhf(gv) + sigf(fv) * c[v];
            float hn = sigf(ov) * tanhf(cn);
            bool live = (t < Lr[v]);                   // freeze past sequence end
            c[v]      = live ? cn : c[v];
            float hv  = live ? hn : hprev[v];
            hprev[v]  = hv;
            out[((size_t)t * BATCH + m) * HID + colh] = hv;
            hdst[m * HID + colh] = f2bf(hv);
        }

        // ---- grid-wide barrier (monotonic counter, reset by lstm_conv) ----
        __threadfence();
        __syncthreads();
        if (threadIdx.x == 0) {
            atomicAdd(bar, 1ull);
            unsigned long long tgt = (unsigned long long)REC_BLOCKS * (t + 1);
            while (*((volatile unsigned long long*)bar) < tgt)
                __builtin_amdgcn_s_sleep(8);
        }
        __syncthreads();
        __threadfence();
    }
}

// ---------------------------------------------------------------- launch ----
extern "C" void kernel_launch(void* const* d_in, const int* in_sizes, int n_in,
                              void* d_out, int out_size, void* d_ws, size_t ws_size,
                              hipStream_t stream) {
    const float* x    = (const float*)d_in[0];   // [256, 64, 1024]
    const float* h0   = (const float*)d_in[1];   // [64, 1024]
    const float* c0   = (const float*)d_in[2];   // [64, 1024]
    const float* W    = (const float*)d_in[3];   // [2048, 4096]
    const float* bias = (const float*)d_in[4];   // [4096]
    const int*   L    = (const int*)d_in[5];     // [64]
    float* out = (float*)d_out;                  // [256, 64, 1024]

    char* ws = (char*)d_ws;
    const size_t WT_OFF  = 0;
    const size_t G_OFF   = WT_OFF + (size_t)N4H * KTOT * 2;           // 16 MiB
    const size_t HB_OFF  = G_OFF + (size_t)MROWS * N4H * 4;           // +256 MiB
    const size_t BAR_OFF = HB_OFF + (size_t)2 * BATCH * HID * 2;      // +256 KiB
    const size_t XBF_OFF = BAR_OFF + 64;
    unsigned short*     Wt   = (unsigned short*)(ws + WT_OFF);
    float*              G    = (float*)(ws + G_OFF);
    unsigned short*     hbuf = (unsigned short*)(ws + HB_OFF);
    unsigned long long* bar  = (unsigned long long*)(ws + BAR_OFF);
    unsigned short*     xbf  = (unsigned short*)(ws + XBF_OFF);

    // 1) bf16 conversions: x, h0 (coalesced) + barrier reset; W^T via LDS tiles
    lstm_conv<<<(MROWS * 1024 / 4) / 256, 256, 0, stream>>>(x, h0, xbf, hbuf, bar);
    lstm_wt<<<dim3(N4H / 64, KTOT / 64), 256, 0, stream>>>(W, Wt);
    // 2) time-parallel pre-gates: G = x @ Wx + b   (bf16 WMMA, f32 acc)
    lstm_xw<<<dim3(N4H / 64, MROWS / 256), 256, 0, stream>>>(xbf, Wt, bias, G);
    // 3) persistent recurrence over 256 steps
    lstm_rec<<<REC_BLOCKS, 256, REC_LDS, stream>>>(G, h0, c0, L, Wt, hbuf, out, bar);
}
